// GCN_4234837753913
// MI455X (gfx1250) — compile-verified, matrix operand order checked by reading, not run on hardware
//
#include <hip/hip_runtime.h>
#include <math.h>

// ---------------------------------------------------------------------------
// GCN (3x GCNConv + MLP + log_softmax) for MI455X / gfx1250, wave32 + WMMA.
// ---------------------------------------------------------------------------

typedef __attribute__((ext_vector_type(16))) _Float16 v16h;
typedef __attribute__((ext_vector_type(8)))  float    v8f;

#define F_IN      256
#define HID       16
#define N_CLS     10
#define NEG_SLOPE 0.02f

// element j of a fragment maps to K = kc + 16*(j>>3) + 8*hi + (j&7)
static __device__ __forceinline__ int kmap(int j, int hi) {
  return ((j >> 3) << 4) + (hi << 3) + (j & 7);
}

// B fragment (K x Ncols row-major weight): lane holds column n.
static __device__ __forceinline__ v16h load_bfrag(const float* __restrict__ W, int ld,
                                                  int n, int hi, int kc, int Kmax, int Nmax) {
  v16h f;
#pragma unroll
  for (int j = 0; j < 16; ++j) {
    int K = kc + kmap(j, hi);
    float v = (K < Kmax && n < Nmax) ? W[K * ld + n] : 0.0f;
    f[j] = (_Float16)v;
  }
  return f;
}

// Generic A fragment from a row-major f32 row pointer, with K zero-padding.
static __device__ __forceinline__ v16h load_afrag(const float* __restrict__ row,
                                                  int hi, int kc, int Kmax) {
  v16h f;
#pragma unroll
  for (int j = 0; j < 16; ++j) {
    int K = kc + kmap(j, hi);
    float v = (K < Kmax) ? row[K] : 0.0f;
    f[j] = (_Float16)v;
  }
  return f;
}

// Vectorized A fragment: K-chunk fully in-bounds (row stride >= kc+32).
static __device__ __forceinline__ v16h load_afrag_vec(const float* __restrict__ row,
                                                      int hi, int kc) {
  const float4* p0 = (const float4*)(row + kc + 8 * hi);
  const float4* p1 = (const float4*)(row + kc + 16 + 8 * hi);
  float4 q0 = p0[0], q1 = p0[1], q2 = p1[0], q3 = p1[1];
  v16h f;
  f[0]  = (_Float16)q0.x; f[1]  = (_Float16)q0.y; f[2]  = (_Float16)q0.z; f[3]  = (_Float16)q0.w;
  f[4]  = (_Float16)q1.x; f[5]  = (_Float16)q1.y; f[6]  = (_Float16)q1.z; f[7]  = (_Float16)q1.w;
  f[8]  = (_Float16)q2.x; f[9]  = (_Float16)q2.y; f[10] = (_Float16)q2.z; f[11] = (_Float16)q2.w;
  f[12] = (_Float16)q3.x; f[13] = (_Float16)q3.y; f[14] = (_Float16)q3.z; f[15] = (_Float16)q3.w;
  return f;
}

// Vectorized A fragment for a 16-wide f32 row, zero-padded to K=32.
static __device__ __forceinline__ v16h load_afrag16(const float* __restrict__ row, int hi) {
  const float4* p = (const float4*)(row + 8 * hi);
  float4 q0 = p[0], q1 = p[1];
  v16h f = {};
  f[0] = (_Float16)q0.x; f[1] = (_Float16)q0.y; f[2] = (_Float16)q0.z; f[3] = (_Float16)q0.w;
  f[4] = (_Float16)q1.x; f[5] = (_Float16)q1.y; f[6] = (_Float16)q1.z; f[7] = (_Float16)q1.w;
  return f;  // elements 8..15 stay zero (K=16..31 pad)
}

// ---------------------------------------------------------------------------
// Degree / normalization kernels
// ---------------------------------------------------------------------------

__global__ void deg_init_kernel(float* __restrict__ deg, int n) {
  int i = blockIdx.x * blockDim.x + threadIdx.x;
  if (i < n) deg[i] = 1.0f;  // self-loop contribution
}

__global__ void deg_edge_kernel(const int* __restrict__ cols, float* __restrict__ deg, int ne) {
  int e = blockIdx.x * blockDim.x + threadIdx.x;
  if (e < ne) atomicAdd(&deg[cols[e]], 1.0f);
}

__global__ void dinv_kernel(float* __restrict__ deg, int n) {
  int i = blockIdx.x * blockDim.x + threadIdx.x;
  if (i < n) deg[i] = rsqrtf(deg[i]);  // deg >= 1 always
}

// ---------------------------------------------------------------------------
// GEMM1: t[N x 16] = x[N x 256] @ W1[256 x 16], WMMA f16->f32
// ---------------------------------------------------------------------------

__global__ void gemm1_kernel(const float* __restrict__ x, const float* __restrict__ W1,
                             float* __restrict__ out, int nrows) {
  __shared__ float sW[F_IN * HID];  // 16 KB
  for (int i = threadIdx.x; i < F_IN * HID; i += blockDim.x) sW[i] = W1[i];
  __syncthreads();

  int wave = threadIdx.x >> 5;
  int lane = threadIdx.x & 31;
  int tile = blockIdx.x * 8 + wave;
  if (tile * 16 >= nrows) return;

  int m0 = tile * 16;
  int m  = m0 + (lane & 15);
  int hi = lane >> 4;
  int n  = lane & 15;
  const float* row = x + (size_t)m * F_IN;

  v8f c = {};
#pragma unroll
  for (int kc = 0; kc < F_IN; kc += 32) {
    v16h a = load_afrag_vec(row, hi, kc);
    v16h b = load_bfrag(sW, HID, n, hi, kc, F_IN, HID);
    c = __builtin_amdgcn_wmma_f32_16x16x32_f16(false, a, false, b, (short)0, c, false, false);
  }
#pragma unroll
  for (int r = 0; r < 8; ++r)
    out[(size_t)(m0 + r + 8 * hi) * HID + n] = c[r];
}

// ---------------------------------------------------------------------------
// Small GEMM: t[N x 16] = h[N x 16] @ W[16 x 16] (K padded to 32)
// ---------------------------------------------------------------------------

__global__ void gemm16_kernel(const float* __restrict__ h, const float* __restrict__ W,
                              float* __restrict__ out, int nrows) {
  int wave = threadIdx.x >> 5;
  int lane = threadIdx.x & 31;
  int tile = blockIdx.x * 8 + wave;
  if (tile * 16 >= nrows) return;

  int m0 = tile * 16;
  int m  = m0 + (lane & 15);
  int hi = lane >> 4;
  int n  = lane & 15;

  v16h a = load_afrag16(h + (size_t)m * HID, hi);
  v16h b = load_bfrag(W, HID, n, hi, 0, HID, HID);
  v8f  c = {};
  c = __builtin_amdgcn_wmma_f32_16x16x32_f16(false, a, false, b, (short)0, c, false, false);
#pragma unroll
  for (int r = 0; r < 8; ++r)
    out[(size_t)(m0 + r + 8 * hi) * HID + n] = c[r];
}

// ---------------------------------------------------------------------------
// Aggregation: out[i] = dinv[i]^2 * t[i] ; out[col] += dinv[row]*dinv[col]*t[row]
// ---------------------------------------------------------------------------

__global__ void self_init_kernel(const float* __restrict__ t, const float* __restrict__ dinv,
                                 float* __restrict__ out, int n) {
  int i = blockIdx.x * blockDim.x + threadIdx.x;
  if (i >= n * HID) return;
  float d = dinv[i >> 4];
  out[i] = d * d * t[i];
}

__global__ void edge_scatter_kernel(const float* __restrict__ t, const float* __restrict__ dinv,
                                    const int* __restrict__ rows, const int* __restrict__ cols,
                                    float* __restrict__ out, int ne) {
  int e = blockIdx.x * blockDim.x + threadIdx.x;
  if (e >= ne) return;
  int r = rows[e], c = cols[e];
  float nrm = dinv[r] * dinv[c];
  const float4* src = (const float4*)(t + (size_t)r * HID);
  float4 v0 = src[0], v1 = src[1], v2 = src[2], v3 = src[3];
  float* dst = out + (size_t)c * HID;
  atomicAdd(dst +  0, nrm * v0.x); atomicAdd(dst +  1, nrm * v0.y);
  atomicAdd(dst +  2, nrm * v0.z); atomicAdd(dst +  3, nrm * v0.w);
  atomicAdd(dst +  4, nrm * v1.x); atomicAdd(dst +  5, nrm * v1.y);
  atomicAdd(dst +  6, nrm * v1.z); atomicAdd(dst +  7, nrm * v1.w);
  atomicAdd(dst +  8, nrm * v2.x); atomicAdd(dst +  9, nrm * v2.y);
  atomicAdd(dst + 10, nrm * v2.z); atomicAdd(dst + 11, nrm * v2.w);
  atomicAdd(dst + 12, nrm * v3.x); atomicAdd(dst + 13, nrm * v3.y);
  atomicAdd(dst + 14, nrm * v3.z); atomicAdd(dst + 15, nrm * v3.w);
}

__global__ void bias_act_kernel(float* __restrict__ h, const float* __restrict__ b,
                                int nelem, int do_relu) {
  int i = blockIdx.x * blockDim.x + threadIdx.x;
  if (i >= nelem) return;
  float v = h[i] + b[i & (HID - 1)];
  if (do_relu) v = fmaxf(v, 0.0f);
  h[i] = v;
}

// ---------------------------------------------------------------------------
// Fused MLP [16 -> 64 -> 16 -> 10] + log_softmax, one 16-row tile per wave.
// ---------------------------------------------------------------------------

__global__ void mlp_kernel(const float* __restrict__ h,
                           const float* __restrict__ Wm1, const float* __restrict__ bm1,
                           const float* __restrict__ Wm2, const float* __restrict__ bm2,
                           const float* __restrict__ Wm3, const float* __restrict__ bm3,
                           float* __restrict__ out, int nrows) {
  __shared__ float sU[8][16 * 64];  // 32 KB: per-wave 16x64 hidden
  __shared__ float sV[8][16 * 16];  // 8 KB
  __shared__ float sL[8][16 * 16];  // 8 KB: logits

  int wave = threadIdx.x >> 5;
  int lane = threadIdx.x & 31;
  int tile = blockIdx.x * 8 + wave;
  if (tile * 16 >= nrows) return;

  int m0 = tile * 16;
  int ml = lane & 15;
  int hi = lane >> 4;
  int n  = lane & 15;

  float* U = sU[wave];
  float* V = sV[wave];
  float* L = sL[wave];

  // ---- layer 1: 16 -> 64, leaky_relu ----
  v16h a1 = load_afrag16(h + (size_t)(m0 + ml) * HID, hi);
#pragma unroll
  for (int nb = 0; nb < 4; ++nb) {
    v16h b = load_bfrag(Wm1, 64, nb * 16 + n, hi, 0, HID, 64);
    v8f  c = {};
    c = __builtin_amdgcn_wmma_f32_16x16x32_f16(false, a1, false, b, (short)0, c, false, false);
    float bias = bm1[nb * 16 + n];
#pragma unroll
    for (int r = 0; r < 8; ++r) {
      float v = c[r] + bias;
      v = (v > 0.0f) ? v : v * NEG_SLOPE;
      U[(r + 8 * hi) * 64 + nb * 16 + n] = v;
    }
  }

  // ---- layer 2: 64 -> 16, leaky_relu (A staged through LDS) ----
  v8f c2 = {};
#pragma unroll
  for (int kc = 0; kc < 64; kc += 32) {
    v16h a = load_afrag(U + ml * 64, hi, kc, 64);
    v16h b = load_bfrag(Wm2, HID, n, hi, kc, 64, HID);
    c2 = __builtin_amdgcn_wmma_f32_16x16x32_f16(false, a, false, b, (short)0, c2, false, false);
  }
  {
    float bias = bm2[n];
#pragma unroll
    for (int r = 0; r < 8; ++r) {
      float v = c2[r] + bias;
      v = (v > 0.0f) ? v : v * NEG_SLOPE;
      V[(r + 8 * hi) * 16 + n] = v;
    }
  }

  // ---- layer 3: 16 -> 10 (N padded to 16, K padded to 32) ----
  v16h a3 = load_afrag(V + ml * 16, hi, 0, HID);
  v16h b3 = load_bfrag(Wm3, N_CLS, n, hi, 0, HID, N_CLS);
  v8f  c3 = {};
  c3 = __builtin_amdgcn_wmma_f32_16x16x32_f16(false, a3, false, b3, (short)0, c3, false, false);
  float bias3 = (n < N_CLS) ? bm3[n] : 0.0f;
#pragma unroll
  for (int r = 0; r < 8; ++r)
    L[(r + 8 * hi) * 16 + n] = c3[r] + bias3;

  // ---- log_softmax over 10 classes, one row per lane (lanes 0..15) ----
  if (lane < 16) {
    const float* lrow = L + lane * 16;
    float mx = lrow[0];
#pragma unroll
    for (int c = 1; c < N_CLS; ++c) mx = fmaxf(mx, lrow[c]);
    float s = 0.0f;
#pragma unroll
    for (int c = 0; c < N_CLS; ++c) s += expf(lrow[c] - mx);
    float lse = mx + logf(s);
    float* orow = out + (size_t)(m0 + lane) * N_CLS;
#pragma unroll
    for (int c = 0; c < N_CLS; ++c) orow[c] = lrow[c] - lse;
  }
}

// ---------------------------------------------------------------------------
// Host launch
// ---------------------------------------------------------------------------

extern "C" void kernel_launch(void* const* d_in, const int* in_sizes, int n_in,
                              void* d_out, int out_size, void* d_ws, size_t ws_size,
                              hipStream_t stream) {
  const float* x   = (const float*)d_in[0];
  const int*   ei  = (const int*)d_in[1];   // [2, E] int32
  const float* W1  = (const float*)d_in[2];
  const float* b1  = (const float*)d_in[3];
  const float* W2  = (const float*)d_in[4];
  const float* b2  = (const float*)d_in[5];
  const float* W3  = (const float*)d_in[6];
  const float* b3  = (const float*)d_in[7];
  const float* Wm1 = (const float*)d_in[8];
  const float* bm1 = (const float*)d_in[9];
  const float* Wm2 = (const float*)d_in[10];
  const float* bm2 = (const float*)d_in[11];
  const float* Wm3 = (const float*)d_in[12];
  const float* bm3 = (const float*)d_in[13];
  float* out = (float*)d_out;

  const int N = in_sizes[0] / F_IN;
  const int E = in_sizes[1] / 2;
  const int* rows = ei;
  const int* cols = ei + E;

  float* dinv = (float*)d_ws;                 // N
  float* bufA = dinv + N;                     // N*16 (GEMM outputs)
  float* bufB = bufA + (size_t)N * HID;       // N*16 (aggregated activations)

  const int tN    = (N + 255) / 256;
  const int tE    = (E + 255) / 256;
  const int tNF   = (N * HID + 255) / 256;
  const int tiles = (N + 15) / 16;
  const int tTile = (tiles + 7) / 8;

  // degrees + symmetric normalization (layer-invariant)
  deg_init_kernel<<<tN, 256, 0, stream>>>(dinv, N);
  deg_edge_kernel<<<tE, 256, 0, stream>>>(cols, dinv, E);
  dinv_kernel<<<tN, 256, 0, stream>>>(dinv, N);

  // layer 1
  gemm1_kernel<<<tTile, 256, 0, stream>>>(x, W1, bufA, N);
  self_init_kernel<<<tNF, 256, 0, stream>>>(bufA, dinv, bufB, N);
  edge_scatter_kernel<<<tE, 256, 0, stream>>>(bufA, dinv, rows, cols, bufB, E);
  bias_act_kernel<<<tNF, 256, 0, stream>>>(bufB, b1, N * HID, 1);

  // layer 2
  gemm16_kernel<<<tTile, 256, 0, stream>>>(bufB, W2, bufA, N);
  self_init_kernel<<<tNF, 256, 0, stream>>>(bufA, dinv, bufB, N);
  edge_scatter_kernel<<<tE, 256, 0, stream>>>(bufA, dinv, rows, cols, bufB, E);
  bias_act_kernel<<<tNF, 256, 0, stream>>>(bufB, b2, N * HID, 1);

  // layer 3 (no relu)
  gemm16_kernel<<<tTile, 256, 0, stream>>>(bufB, W3, bufA, N);
  self_init_kernel<<<tNF, 256, 0, stream>>>(bufA, dinv, bufB, N);
  edge_scatter_kernel<<<tE, 256, 0, stream>>>(bufA, dinv, rows, cols, bufB, E);
  bias_act_kernel<<<tNF, 256, 0, stream>>>(bufB, b3, N * HID, 0);

  // fused MLP + log_softmax
  mlp_kernel<<<tTile, 256, 0, stream>>>(bufB, Wm1, bm1, Wm2, bm2, Wm3, bm3, out, N);
}
